// Net_59700045415107
// MI455X (gfx1250) — matrix operand order repaired and checked by reading, not compiled, not run on hardware
//
#include <hip/hip_runtime.h>
#include <hip/hip_bf16.h>
#include <stdint.h>

// ---------------- problem constants ----------------
#define LSEQ   2048
#define HDIM   256
#define G4     1024          // 4*H
#define VOCAB  32000
#define NOUT   ((size_t)LSEQ * (size_t)VOCAB)

// ---------------- WMMA fragment types ----------------
typedef float  v8f  __attribute__((ext_vector_type(8)));
typedef float  v2f  __attribute__((ext_vector_type(2)));
typedef __bf16 v16bf __attribute__((ext_vector_type(16)));
typedef unsigned v4u __attribute__((ext_vector_type(4)));
typedef int      v8i __attribute__((ext_vector_type(8)));
typedef int      v4i __attribute__((ext_vector_type(4)));

union Frag16 {           // 16x32 bf16 A-frag or 32x16 bf16 B-frag (8 VGPRs/lane)
    v16bf    v;
    unsigned u[8];
    uint4    q[2];
};

#if __has_builtin(__builtin_amdgcn_wmma_f32_16x16x4_f32)
#define HAVE_WMMA_F32 1
#else
#define HAVE_WMMA_F32 0
#endif

#if __has_builtin(__builtin_amdgcn_tensor_load_to_lds) && \
    __has_builtin(__builtin_amdgcn_s_wait_tensorcnt)
#define HAVE_TDM 1
#else
#define HAVE_TDM 0
#endif

// ---------------- helpers ----------------
__device__ __forceinline__ unsigned short f2bf(float f) {
    unsigned u = __float_as_uint(f);
    u += 0x7fffu + ((u >> 16) & 1u);       // round-to-nearest-even
    return (unsigned short)(u >> 16);
}
__device__ __forceinline__ float sigm(float x) {
    return 1.0f / (1.0f + __expf(-x));
}

// ---------------- kernel 1: f32 -> bf16 convert (W_lin) ----------------
__global__ void k_convert_bf16(const float* __restrict__ src,
                               unsigned short* __restrict__ dst, int n) {
    int i = blockIdx.x * blockDim.x + threadIdx.x;
    if (i < n) dst[i] = f2bf(src[i]);
}

// ---------------- kernel 2: pack W_hh into k-major bf16 pairs ----------------
// wpk[k2*1024 + j] = { bf16(W_hh[j][2k2]), bf16(W_hh[j][2k2+1]) }
__global__ void k_pack_whh(const float* __restrict__ whh,
                           unsigned* __restrict__ wpk) {
    int idx = blockIdx.x * blockDim.x + threadIdx.x;   // 0 .. 128*1024-1
    if (idx >= 128 * G4) return;
    int j  = idx & (G4 - 1);
    int k2 = idx >> 10;
    unsigned lo = f2bf(whh[(size_t)j * HDIM + 2 * k2]);
    unsigned hi = f2bf(whh[(size_t)j * HDIM + 2 * k2 + 1]);
    wpk[idx] = lo | (hi << 16);
}

// ---------------- kernel 3: x_proj = gather(emb) @ W_ih^T + b_ih + b_hh ------
// fp32 WMMA 16x16x4. Grid (1024/128, 2048/64) = (8, 32), 256 threads (8 waves).
// Wave (wm,wn): 16(M) x 64(N) strip, 4 n-subtiles per wave.
__global__ __launch_bounds__(256)
void k_xproj(const int* __restrict__ inp, const float* __restrict__ emb,
             const float* __restrict__ Wih, const float* __restrict__ bih,
             const float* __restrict__ bhh, float* __restrict__ xp) {
#if HAVE_WMMA_F32
    const int tid  = threadIdx.x;
    const int lane = tid & 31, w = tid >> 5;
    const int wm = w & 3, wn = w >> 2;
    const int m0 = blockIdx.y * 64 + wm * 16;
    const int n0 = blockIdx.x * 128 + wn * 64;
    const int l16 = lane & 15, kh = lane >> 4;
    const int m = m0 + l16;

    const float* arow = emb + (size_t)inp[m] * HDIM;       // gathered A row
    const float* brow = Wih + (size_t)(n0 + l16) * HDIM;   // B: W_ih[n][k]

    v8f acc[4] = {};
    for (int kk = 0; kk < HDIM; kk += 4) {
        // f32 A 16x4: lanes 0-15 -> K = kk+0,1 ; lanes 16-31 -> K = kk+2,3
        v2f a = *(const v2f*)(arow + kk + 2 * kh);
        #pragma unroll
        for (int nt = 0; nt < 4; ++nt) {
            v2f b = *(const v2f*)(brow + (size_t)nt * 16 * HDIM + kk + 2 * kh);
            acc[nt] = __builtin_amdgcn_wmma_f32_16x16x4_f32(
                false, a, false, b, (short)0, acc[nt], false, false);
        }
    }
    #pragma unroll
    for (int nt = 0; nt < 4; ++nt) {
        int n = n0 + nt * 16 + l16;
        float bias = bih[n] + bhh[n];
        #pragma unroll
        for (int v = 0; v < 8; ++v) {
            int mm = m0 + v + 8 * kh;                 // C/D layout: M = v + 8*(lane>=16)
            xp[(size_t)mm * G4 + n] = acc[nt][v] + bias;
        }
    }
#else
    // scalar fallback (same grid/tile)
    const int tid = threadIdx.x;
    const int m0 = blockIdx.y * 64;
    const int n0 = blockIdx.x * 128;
    for (int o = tid; o < 64 * 128; o += 256) {
        int mm = m0 + o / 128, n = n0 + (o & 127);
        const float* ar = emb + (size_t)inp[mm] * HDIM;
        const float* br = Wih + (size_t)n * HDIM;
        float s = bih[n] + bhh[n];
        for (int k = 0; k < HDIM; ++k) s = fmaf(ar[k], br[k], s);
        xp[(size_t)mm * G4 + n] = s;
    }
#endif
}

// ---------------- kernel 4: sequential LSTM scan (single WGP) ----------------
// 1024 threads: thread j owns gate j. h broadcast via LDS; weights streamed
// coalesced from L2 (k-major bf16 pairs). Writes hs (bf16) and hL/cL tail.
__global__ __launch_bounds__(1024)
void k_scan(const float* __restrict__ xp, const unsigned* __restrict__ wpk,
            const float* __restrict__ h0, const float* __restrict__ c0,
            unsigned short* __restrict__ hsb, float* __restrict__ out) {
    __shared__ float h_s[HDIM];
    __shared__ float g_s[G4];
    const int j = threadIdx.x;
    float c = 0.0f;
    if (j < HDIM) { h_s[j] = h0[j]; c = c0[j]; }
    __syncthreads();

    for (int t = 0; t < LSEQ; ++t) {
        float g = xp[(size_t)t * G4 + j];
        const unsigned* wc = wpk + j;
        #pragma unroll 8
        for (int k2 = 0; k2 < 128; ++k2) {
            unsigned p = wc[k2 << 10];                        // coalesced across lanes
            float wlo = __uint_as_float(p << 16);             // bf16 -> f32 = shl 16
            float whi = __uint_as_float(p & 0xffff0000u);
            g = fmaf(wlo, h_s[2 * k2],     g);                // LDS broadcast reads
            g = fmaf(whi, h_s[2 * k2 + 1], g);
        }
        g_s[j] = g;
        __syncthreads();
        if (j < HDIM) {
            float i_ = sigm(g_s[j]);
            float f_ = sigm(g_s[HDIM + j]);
            float gg = tanhf(g_s[2 * HDIM + j]);
            float o_ = sigm(g_s[3 * HDIM + j]);
            c = fmaf(f_, c, i_ * gg);
            float hn = o_ * tanhf(c);
            h_s[j] = hn;
            hsb[(size_t)t * HDIM + j] = f2bf(hn);
        }
        __syncthreads();
    }
    if (j < HDIM) {
        out[NOUT + j]        = h_s[j];   // hL
        out[NOUT + HDIM + j] = c;        // cL
    }
}

// ---------------- kernel 5: logits = relu(hs @ W_lin^T + b) ------------------
// bf16 WMMA 16x16x32. Grid (32000/128, 2048/64) = (250, 32), 256 threads.
// B operand (W_lin strip) staged into LDS by the Tensor Data Mover in two
// 128-K chunks with TDM padding (row stride 272B) so ds_load_b128 B-fragment
// reads are bank-conflict-free. A-frags are two contiguous global_load_b128s.
#define BK 128
#define BROW 136   // LDS row stride in bf16 elements (272 B = 64 dwords + 4 pad)

__global__ __launch_bounds__(256)
void k_gemm_vocab(const unsigned short* __restrict__ hsb,
                  const unsigned short* __restrict__ wlb,
                  const float* __restrict__ blin, float* __restrict__ out) {
    const int tid  = threadIdx.x;
    const int lane = tid & 31, w = tid >> 5;
    const int wm = w & 3, wn = w >> 2;
    const int m0    = blockIdx.y * 64 + wm * 16;
    const int n0blk = blockIdx.x * 128;
    const int l16 = lane & 15, kh = lane >> 4;

    const unsigned short* arow = hsb + (size_t)(m0 + l16) * HDIM;
    v8f acc[4] = {};

#if HAVE_TDM
    __shared__ unsigned short b_lds[128 * BROW];          // 34,816 bytes
    const unsigned lds_base = (unsigned)(uintptr_t)&b_lds[0];  // LDS byte offset
    const unsigned short* bb = b_lds + (size_t)(wn * 64 + l16) * BROW + kh * 16;

    for (int c = 0; c < 2; ++c) {
        const int kkc = c * BK;
        __syncthreads();                                  // prior chunk consumed
        if (w == 0) {
            // Tensor DMA descriptor (D#), per CDNA5 ISA ch.8:
            // 2D tile: tile_dim0 = 128 K-elems (contig), tile_dim1 = 128 n-rows,
            // data_size = 2B, tensor_dim0_stride = 256 elems, pad 4 dwords per
            // 64-dword interval -> LDS row stride 272 B.
            unsigned long long ga =
                (unsigned long long)(uintptr_t)(wlb + (size_t)n0blk * HDIM + kkc);
            v4u g0 = { 1u,                                  // count=1, user D#
                       lds_base,                            // lds_addr
                       (unsigned)ga,                        // global_addr[31:0]
                       (unsigned)((ga >> 32) & 0x01ffffffu) // global_addr[56:32]
                         | (2u << 30) };                    // type=2 (image)
            v8i g1 = { (int)0x07510000u,    // data_size=2B, pad_en, ivl=64dw, amt=4dw
                       (int)(128u << 16),   // tensor_dim0 = 128
                       (int)(128u << 16),   // tensor_dim1 = 128
                       (int)(128u << 16),   // tile_dim0   = 128
                       128,                 // tile_dim1   = 128 (tile_dim2 = 0)
                       256,                 // tensor_dim0_stride = 256
                       0, 0 };
            v4i gz = { 0, 0, 0, 0 };
#if __clang_major__ >= 23
            v8i gz8 = { 0, 0, 0, 0, 0, 0, 0, 0 };
            __builtin_amdgcn_tensor_load_to_lds(g0, g1, gz, gz, gz8, 0);
#else
            __builtin_amdgcn_tensor_load_to_lds(g0, g1, gz, gz, 0);
#endif
            __builtin_amdgcn_s_wait_tensorcnt(0);
        }
        __syncthreads();                                  // LDS tile visible to all

        for (int kk = 0; kk < BK; kk += 32) {
            Frag16 a;   // A 16x32: two contiguous B128s (K=kk+16kh.., kk+16kh+8..)
            a.q[0] = *(const uint4*)(arow + kkc + kk + 16 * kh);
            a.q[1] = *(const uint4*)(arow + kkc + kk + 16 * kh + 8);
            #pragma unroll
            for (int nt = 0; nt < 4; ++nt) {
                Frag16 b;   // B 32x16 from LDS: 16 contiguous K per lane-half
                const uint4* p = (const uint4*)(bb + (size_t)nt * 16 * BROW + kk);
                b.q[0] = p[0];
                b.q[1] = p[1];
                acc[nt] = __builtin_amdgcn_wmma_f32_16x16x32_bf16(
                    false, a.v, false, b.v, (short)0, acc[nt], false, false);
            }
        }
    }
#else
    const unsigned short* brow = wlb + (size_t)(n0blk + wn * 64 + l16) * HDIM + kh * 16;
    for (int kk = 0; kk < HDIM; kk += 32) {
        __builtin_prefetch((const void*)(brow + kk + 32), 0, 0);
        Frag16 a;
        a.q[0] = *(const uint4*)(arow + kk + 16 * kh);
        a.q[1] = *(const uint4*)(arow + kk + 16 * kh + 8);
        #pragma unroll
        for (int nt = 0; nt < 4; ++nt) {
            Frag16 b;
            const uint4* p = (const uint4*)(brow + (size_t)nt * 16 * HDIM + kk);
            b.q[0] = p[0];
            b.q[1] = p[1];
            acc[nt] = __builtin_amdgcn_wmma_f32_16x16x32_bf16(
                false, a.v, false, b.v, (short)0, acc[nt], false, false);
        }
    }
#endif

    #pragma unroll
    for (int nt = 0; nt < 4; ++nt) {
        int n = n0blk + wn * 64 + nt * 16 + l16;
        float bias = blin[n];
        #pragma unroll
        for (int v = 0; v < 8; ++v) {
            int mm = m0 + v + 8 * kh;
            float val = acc[nt][v] + bias;
            out[(size_t)mm * VOCAB + n] = fmaxf(val, 0.0f);   // fused ReLU
        }
    }
}

// ---------------- kernel 6: per-row max & log-sum-exp ------------------------
__global__ __launch_bounds__(256)
void k_rowstats(const float* __restrict__ out, float* __restrict__ lse) {
    __shared__ float red[256];
    const int tid = threadIdx.x;
    const float* row = out + (size_t)blockIdx.x * VOCAB;

    float m = -INFINITY;
    for (int i = tid; i < VOCAB; i += 256) m = fmaxf(m, row[i]);
    red[tid] = m; __syncthreads();
    for (int s = 128; s > 0; s >>= 1) {
        if (tid < s) red[tid] = fmaxf(red[tid], red[tid + s]);
        __syncthreads();
    }
    m = red[0]; __syncthreads();

    float s_ = 0.0f;
    for (int i = tid; i < VOCAB; i += 256) s_ += __expf(row[i] - m);
    red[tid] = s_; __syncthreads();
    for (int s = 128; s > 0; s >>= 1) {
        if (tid < s) red[tid] += red[tid + s];
        __syncthreads();
    }
    if (tid == 0) lse[blockIdx.x] = m + __logf(red[0]);
}

// ---------------- kernel 7: out = out - lse[row] (vectorized) ----------------
__global__ __launch_bounds__(256)
void k_sub(float* __restrict__ out, const float* __restrict__ lse) {
    int i4 = blockIdx.x * 256 + threadIdx.x;          // 16,384,000 float4s
    float4* o4 = (float4*)out;
    float l = lse[i4 / (VOCAB / 4)];
    float4 v = o4[i4];
    v.x -= l; v.y -= l; v.z -= l; v.w -= l;
    o4[i4] = v;
}

// ---------------- launcher ----------------
extern "C" void kernel_launch(void* const* d_in, const int* in_sizes, int n_in,
                              void* d_out, int out_size, void* d_ws, size_t ws_size,
                              hipStream_t stream) {
    (void)in_sizes; (void)n_in; (void)out_size; (void)ws_size;
    const int*   input = (const int*)  d_in[0];
    const float* h0    = (const float*)d_in[1];
    const float* c0    = (const float*)d_in[2];
    const float* emb   = (const float*)d_in[3];
    const float* W_ih  = (const float*)d_in[4];
    const float* W_hh  = (const float*)d_in[5];
    const float* b_ih  = (const float*)d_in[6];
    const float* b_hh  = (const float*)d_in[7];
    const float* W_lin = (const float*)d_in[8];
    const float* b_lin = (const float*)d_in[9];
    float* out = (float*)d_out;

    char* ws = (char*)d_ws;
    size_t off = 0;
    float*          xproj   = (float*)(ws + off);          off += (size_t)LSEQ * G4 * 4;      // 8 MB
    unsigned*       whh_pk  = (unsigned*)(ws + off);       off += (size_t)128 * G4 * 4;       // 0.5 MB
    unsigned short* wlin_bf = (unsigned short*)(ws + off); off += (size_t)VOCAB * HDIM * 2;   // 16 MB
    unsigned short* hs_bf   = (unsigned short*)(ws + off); off += (size_t)LSEQ * HDIM * 2;    // 1 MB
    float*          lse     = (float*)(ws + off);          off += (size_t)LSEQ * 4;           // 8 KB

    // 1) W_lin f32 -> bf16 (L2-resident operand for the big WMMA GEMM)
    k_convert_bf16<<<(VOCAB * HDIM + 255) / 256, 256, 0, stream>>>(W_lin, wlin_bf, VOCAB * HDIM);
    // 2) W_hh -> k-major packed bf16 pairs (coalesced recurrent matvec)
    k_pack_whh<<<(128 * G4 + 255) / 256, 256, 0, stream>>>(W_hh, whh_pk);
    // 3) input projection GEMM, fp32 WMMA 16x16x4 (gather fused)
    k_xproj<<<dim3(G4 / 128, LSEQ / 64), 256, 0, stream>>>(input, emb, W_ih, b_ih, b_hh, xproj);
    // 4) sequential LSTM scan (single workgroup, LDS broadcast h)
    k_scan<<<1, 1024, 0, stream>>>(xproj, whh_pk, h0, c0, hs_bf, out);
    // 5) vocab GEMM + bias + ReLU, bf16 WMMA 16x16x32, TDM-staged B operand
    k_gemm_vocab<<<dim3(VOCAB / 128, LSEQ / 64), 256, 0, stream>>>(hs_bf, wlin_bf, b_lin, out);
    // 6) row max + log-sum-exp
    k_rowstats<<<LSEQ, 256, 0, stream>>>(out, lse);
    // 7) log_softmax subtraction
    k_sub<<<(int)(NOUT / 4 / 256), 256, 0, stream>>>(out, lse);
}
